// GPT_wrapper_scratch_21303037788838
// MI455X (gfx1250) — matrix operand order repaired and checked by reading, not compiled, not run on hardware
//
#include <hip/hip_runtime.h>

// ---------------------------------------------------------------------------
// GPT-2 forward on gfx1250 (MI455X). bf16 WMMA GEMMs with fp32 accumulate.
// Residual stream fp32; LN / softmax / GELU emit bf16 operands for next GEMM.
// Weights converted fp32->bf16 once into workspace (lm_head transposed so all
// B operands stage with 16-byte vector loads).
// Shape contract (guaranteed by the launches below): M % 128 == 0, K % 32 == 0,
// every operand 16B-aligned with leading dims % 8 == 0; N is the only edge
// dimension (handled with per-thread predicated vector loads, zero fill).
// LDS tiles use stride-40 rows: bank-conflict-free b128 fragment loads.
// ---------------------------------------------------------------------------

typedef __bf16 bf16;
typedef __attribute__((ext_vector_type(16))) __bf16 v16bf;
typedef __attribute__((ext_vector_type(8)))  float  v8f;

#define GPT_L 12
#define GPT_H 16
#define GPT_C 1024
#define GPT_V 50257
#define GPT_B 2
#define GPT_T 1024
#define GPT_D 64   // C / H

// ---------------------------------------------------------------------------
// reductions
// ---------------------------------------------------------------------------
__device__ __forceinline__ float blockReduceSum(float v) {
    __shared__ float sm[8];
    __shared__ float res;
    __syncthreads();
    #pragma unroll
    for (int o = 16; o > 0; o >>= 1) v += __shfl_xor(v, o, 32);
    int lane = threadIdx.x & 31, w = threadIdx.x >> 5;
    if (lane == 0) sm[w] = v;
    __syncthreads();
    if (threadIdx.x == 0) {
        float s = 0.f;
        #pragma unroll
        for (int i = 0; i < 8; ++i) s += sm[i];
        res = s;
    }
    __syncthreads();
    return res;
}

__device__ __forceinline__ float blockReduceMax(float v) {
    __shared__ float sm[8];
    __shared__ float res;
    __syncthreads();
    #pragma unroll
    for (int o = 16; o > 0; o >>= 1) v = fmaxf(v, __shfl_xor(v, o, 32));
    int lane = threadIdx.x & 31, w = threadIdx.x >> 5;
    if (lane == 0) sm[w] = v;
    __syncthreads();
    if (threadIdx.x == 0) {
        float s = -3.4e38f;
        #pragma unroll
        for (int i = 0; i < 8; ++i) s = fmaxf(s, sm[i]);
        res = s;
    }
    __syncthreads();
    return res;
}

__device__ __forceinline__ float gelu_f(float x) {
    const float k = 0.7978845608028654f;
    float x3 = x * x * x;
    return 0.5f * x * (1.0f + tanhf(k * (x + 0.044715f * x3)));
}

// ---------------------------------------------------------------------------
// elementwise kernels
// ---------------------------------------------------------------------------
__global__ __launch_bounds__(256) void f32_to_bf16_kernel(const float* __restrict__ in,
                                                          bf16* __restrict__ out, long n) {
    long i = (long)blockIdx.x * blockDim.x + threadIdx.x;
    long stride = (long)gridDim.x * blockDim.x;
    for (; i < n; i += stride) out[i] = (bf16)in[i];
}

// in: [K][N] f32 row-major  ->  out: [N][K] bf16 row-major
__global__ __launch_bounds__(256) void transpose_cvt_kernel(const float* __restrict__ in,
                                                            bf16* __restrict__ out,
                                                            int K, int N) {
    __shared__ float tile[32][33];
    int nBase = blockIdx.x * 32;
    int kBase = blockIdx.y * 32;
    int tx = threadIdx.x & 31;
    int ty = threadIdx.x >> 5;       // 8 rows of threads
    #pragma unroll
    for (int i = ty; i < 32; i += 8) {
        int k = kBase + i, n = nBase + tx;
        tile[i][tx] = (k < K && n < N) ? in[(long)k * N + n] : 0.f;
    }
    __syncthreads();
    #pragma unroll
    for (int i = ty; i < 32; i += 8) {
        int n = nBase + i, k = kBase + tx;
        if (n < N && k < K) out[(long)n * K + k] = (bf16)tile[tx][i];
    }
}

__global__ __launch_bounds__(256) void embed_kernel(const int* __restrict__ idx,
                                                    const float* __restrict__ wte,
                                                    const float* __restrict__ wpe,
                                                    float* __restrict__ x, int T, int C) {
    int row = blockIdx.x;            // b*T + t
    int tok = idx[row];
    int t = row % T;
    const float* we = wte + (long)tok * C;
    const float* wp = wpe + (long)t * C;
    float* xr = x + (long)row * C;
    for (int i = threadIdx.x; i < C; i += blockDim.x) xr[i] = we[i] + wp[i];
}

__global__ __launch_bounds__(256) void ln_bf16_kernel(const float* __restrict__ x,
                                                      const float* __restrict__ w,
                                                      const float* __restrict__ b,
                                                      bf16* __restrict__ out, int C) {
    int row = blockIdx.x;
    const float* xr = x + (long)row * C;
    bf16* orow = out + (long)row * C;
    float s = 0.f;
    for (int i = threadIdx.x; i < C; i += blockDim.x) s += xr[i];
    float mean = blockReduceSum(s) / (float)C;
    float v = 0.f;
    for (int i = threadIdx.x; i < C; i += blockDim.x) {
        float d = xr[i] - mean; v += d * d;
    }
    float var = blockReduceSum(v) / (float)C;
    float rstd = rsqrtf(var + 1e-5f);
    for (int i = threadIdx.x; i < C; i += blockDim.x)
        orow[i] = (bf16)((xr[i] - mean) * rstd * w[i] + b[i]);
}

__global__ __launch_bounds__(256) void softmax_bf16_kernel(const float* __restrict__ scores,
                                                           bf16* __restrict__ probs, int T) {
    long row = blockIdx.x;           // (b*H + h)*T + t
    const float* srow = scores + row * T;
    bf16* prow = probs + row * T;
    float mx = -3.4e38f;
    for (int i = threadIdx.x; i < T; i += blockDim.x) mx = fmaxf(mx, srow[i]);
    mx = blockReduceMax(mx);
    float loc[4];
    float sum = 0.f;
    #pragma unroll
    for (int j = 0; j < 4; ++j) {
        int i = threadIdx.x + j * 256;
        float e = (i < T) ? __expf(srow[i] - mx) : 0.f;
        loc[j] = e; sum += e;
    }
    sum = blockReduceSum(sum);
    float inv = 1.0f / sum;
    #pragma unroll
    for (int j = 0; j < 4; ++j) {
        int i = threadIdx.x + j * 256;
        if (i < T) prow[i] = (bf16)(loc[j] * inv);
    }
}

// ---------------------------------------------------------------------------
// Generic batched bf16 WMMA GEMM:  C[z] = epi( A[z](MxK) @ B[z](KxN) )
//   A (m,k) at A[m*lda + k]            (k contiguous)
//   B (k,n) at B[k*ldbk + n*ldbn]      KCONTIG ? ldbk==1 : ldbn==1
// Tile 128x128x32, 256 threads = 8 waves, wave tile 32x64 (8 WMMAs / K-step).
// EPI: 0 = f32 out (scale, optional causal mask)
//      1 = f32 out = resid + acc + bias
//      2 = bf16 out = acc + bias
//      3 = bf16 out = gelu(acc + bias)
// Only the (KCONTIG,EPI) combinations actually used are instantiated:
//   (false,1) (false,2) (false,3) (true,0)
// ---------------------------------------------------------------------------
#define BM 128
#define BN 128
#define BK 32
#define LDSS 40   // LDS row stride in elements (80B = 20 banks: conflict-free)

template<bool KCONTIG, int EPI>
__global__ __launch_bounds__(256) void wmma_gemm_kernel(
    const bf16* __restrict__ A, long lda,
    const bf16* __restrict__ Bmat, long ldbk, long ldbn,
    const float* __restrict__ bias,
    const float* __restrict__ resid,
    float* __restrict__ outF, bf16* __restrict__ outB, long ldc,
    int M, int N, int K,
    float scale, int maskCausal,
    int zInner, long aOut, long aIn, long bOut, long bIn, long cOut, long cIn)
{
    int z = blockIdx.z;
    int zo = z / zInner, zi = z % zInner;
    A    += zo * aOut + zi * aIn;
    Bmat += zo * bOut + zi * bIn;
    long coff = zo * cOut + zi * cIn;
    if (outF)  outF  += coff;
    if (outB)  outB  += coff;
    if (resid) resid += coff;

    const int m0 = blockIdx.y * BM;
    const int n0 = blockIdx.x * BN;

    __shared__ bf16 As[BM * LDSS];   // [m][k]
    __shared__ bf16 Bs[BN * LDSS];   // [n][k]  (transposed tile)

    const int tid  = threadIdx.x;
    const int lane = tid & 31;
    const int wave = tid >> 5;
    const int wm0  = (wave & 3) * 32;   // 4 wave rows
    const int wn0  = (wave >> 2) * 64;  // 2 wave cols

    v8f acc[2][4];
    #pragma unroll
    for (int mi = 0; mi < 2; ++mi)
        #pragma unroll
        for (int ni = 0; ni < 4; ++ni)
            acc[mi][ni] = (v8f){0.f,0.f,0.f,0.f,0.f,0.f,0.f,0.f};

    const int half = lane >> 4;      // frag K-half selector
    const int ar   = lane & 15;      // frag row/col within 16

    // per-thread staging coordinates (fixed across K loop)
    const int a_r  = tid >> 1;
    const int a_ks = (tid & 1) * 16;
    const int bk_n  = tid >> 1;           // KCONTIG: one n per thread
    const int bk_ks = (tid & 1) * 16;
    const int bn_k  = tid >> 3;           // !KCONTIG: one k per thread
    const int bn_ns = (tid & 7) * 16;

    const uint4 zero4 = {0u, 0u, 0u, 0u};
    union Frag { v16bf v; uint4 q[2]; };

    for (int k0 = 0; k0 < K; k0 += BK) {
        // ---- stage A: 16 contiguous k per thread, unconditional b128 ----
        {
            const bf16* src = A + (long)(m0 + a_r) * lda + (k0 + a_ks);
            const uint4* s4 = (const uint4*)src;
            uint4 q0 = s4[0];
            uint4 q1 = s4[1];
            bf16* dst = &As[a_r * LDSS + a_ks];
            *(uint4*)dst = q0;
            *((uint4*)dst + 1) = q1;
            if (k0 + BK < K) __builtin_prefetch(src + BK, 0, 0);
        }
        // ---- stage B transposed into Bs[n][k] ----
        if (KCONTIG) {
            int gn = n0 + bk_n;
            uint4 q0 = zero4, q1 = zero4;
            if (gn < N) {                        // per-thread predicate only
                const uint4* s4 = (const uint4*)(Bmat + (long)gn * ldbn + (k0 + bk_ks));
                q0 = s4[0];
                q1 = s4[1];
                if (k0 + BK < K) __builtin_prefetch((const bf16*)s4 + BK, 0, 0);
            }
            bf16* dst = &Bs[bk_n * LDSS + bk_ks];
            *(uint4*)dst = q0;
            *((uint4*)dst + 1) = q1;
        } else {
            int gn = n0 + bn_ns;
            union { uint4 q[2]; bf16 h[16]; } u;
            u.q[0] = zero4; u.q[1] = zero4;
            if (gn + 16 <= N) {                  // N % 16 == 0 in this mode
                const uint4* s4 = (const uint4*)(Bmat + (long)(k0 + bn_k) * ldbk + gn);
                u.q[0] = s4[0];
                u.q[1] = s4[1];
                if (k0 + BK < K) __builtin_prefetch((const bf16*)s4 + (long)BK * ldbk, 0, 0);
            }
            #pragma unroll
            for (int j = 0; j < 16; ++j)
                Bs[(bn_ns + j) * LDSS + bn_k] = u.h[j];
        }
        __syncthreads();

        // ---- fragments: 2 x ds_load_b128 each ----
        Frag afrag[2];
        #pragma unroll
        for (int mi = 0; mi < 2; ++mi) {
            const bf16* p = &As[(wm0 + mi * 16 + ar) * LDSS + half * 8];
            afrag[mi].q[0] = *(const uint4*)p;         // k = half*8 .. +7
            afrag[mi].q[1] = *(const uint4*)(p + 16);  // k = 16+half*8 .. +7
        }
        Frag bfrag[4];
        #pragma unroll
        for (int ni = 0; ni < 4; ++ni) {
            const bf16* p = &Bs[(wn0 + ni * 16 + ar) * LDSS + half * 8];
            bfrag[ni].q[0] = *(const uint4*)p;         // k = half*8 .. +7
            bfrag[ni].q[1] = *(const uint4*)(p + 16);  // k = 16+half*8 .. +7
        }
        // ---- 8 WMMAs ----
        #pragma unroll
        for (int mi = 0; mi < 2; ++mi)
            #pragma unroll
            for (int ni = 0; ni < 4; ++ni)
                acc[mi][ni] = __builtin_amdgcn_wmma_f32_16x16x32_bf16(
                    false, afrag[mi].v, false, bfrag[ni].v,
                    (short)0, acc[mi][ni], false, false);
        __syncthreads();
    }

    // ---- epilogue (C: m = (lane>>4)*8 + r, n = lane&15); only N can edge ----
    const int cn  = lane & 15;
    const int cm0 = (lane >> 4) * 8;
    #pragma unroll
    for (int mi = 0; mi < 2; ++mi) {
        #pragma unroll
        for (int ni = 0; ni < 4; ++ni) {
            int gn = n0 + wn0 + ni * 16 + cn;
            if (gn < N) {
                #pragma unroll
                for (int r = 0; r < 8; ++r) {
                    int gm = m0 + wm0 + mi * 16 + cm0 + r;
                    float v = acc[mi][ni][r] * scale;
                    if (bias) v += bias[gn];
                    if (maskCausal && gn > gm) v = -1e30f;
                    long o = (long)gm * ldc + gn;
                    if (EPI == 0)      outF[o] = v;
                    else if (EPI == 1) outF[o] = resid[o] + v;
                    else if (EPI == 2) outB[o] = (bf16)v;
                    else               outB[o] = (bf16)gelu_f(v);
                }
            }
        }
    }
}

// ---------------------------------------------------------------------------
// host
// ---------------------------------------------------------------------------
extern "C" void kernel_launch(void* const* d_in, const int* in_sizes, int n_in,
                              void* d_out, int out_size, void* d_ws, size_t ws_size,
                              hipStream_t stream) {
    const int   B = GPT_B, T = GPT_T, C = GPT_C, H = GPT_H, D = GPT_D;
    const int   L = GPT_L, V = GPT_V;
    const int   M = B * T;

    const int*   idx      = (const int*)  d_in[0];
    const float* wte      = (const float*)d_in[1];
    const float* wpe      = (const float*)d_in[2];
    const float* ln1_w    = (const float*)d_in[3];
    const float* ln1_b    = (const float*)d_in[4];
    const float* attn_w   = (const float*)d_in[5];
    const float* attn_b   = (const float*)d_in[6];
    const float* proj_w   = (const float*)d_in[7];
    const float* proj_b   = (const float*)d_in[8];
    const float* ln2_w    = (const float*)d_in[9];
    const float* ln2_b    = (const float*)d_in[10];
    const float* fc_w     = (const float*)d_in[11];
    const float* fc_b     = (const float*)d_in[12];
    const float* fcproj_w = (const float*)d_in[13];
    const float* fcproj_b = (const float*)d_in[14];
    const float* lnf_w    = (const float*)d_in[15];
    const float* lnf_b    = (const float*)d_in[16];
    const float* lm_w     = (const float*)d_in[17];
    float* logits = (float*)d_out;

    size_t off = 0;
    auto alloc = [&](size_t bytes) -> void* {
        void* p = (char*)d_ws + off;
        off = (off + bytes + 255) & ~(size_t)255;
        return p;
    };
    bf16* attn_w_bf   = (bf16*)alloc((size_t)L * C * 3 * C * sizeof(bf16));
    bf16* proj_w_bf   = (bf16*)alloc((size_t)L * C * C * sizeof(bf16));
    bf16* fc_w_bf     = (bf16*)alloc((size_t)L * C * 4 * C * sizeof(bf16));
    bf16* fcproj_w_bf = (bf16*)alloc((size_t)L * 4 * C * C * sizeof(bf16));
    bf16* lm_wT_bf    = (bf16*)alloc((size_t)C * V * sizeof(bf16));   // [V][C]
    float* x          = (float*)alloc((size_t)M * C * sizeof(float));
    bf16* ln_out      = (bf16*) alloc((size_t)M * C * sizeof(bf16));
    bf16* qkv         = (bf16*) alloc((size_t)M * 3 * C * sizeof(bf16));
    float* scores     = (float*)alloc((size_t)B * H * T * T * sizeof(float));
    bf16* probs       = (bf16*) alloc((size_t)B * H * T * T * sizeof(bf16));
    bf16* atty        = (bf16*) alloc((size_t)M * C * sizeof(bf16));
    bf16* fc_out      = (bf16*) alloc((size_t)M * 4 * C * sizeof(bf16));
    (void)ws_size; (void)n_in; (void)in_sizes; (void)out_size;

    auto cvt = [&](const float* src, bf16* dst, long n) {
        int grid = (int)((n + 256L * 16 - 1) / (256L * 16));
        if (grid < 1) grid = 1;
        f32_to_bf16_kernel<<<grid, 256, 0, stream>>>(src, dst, n);
    };
    cvt(attn_w,   attn_w_bf,   (long)L * C * 3 * C);
    cvt(proj_w,   proj_w_bf,   (long)L * C * C);
    cvt(fc_w,     fc_w_bf,     (long)L * C * 4 * C);
    cvt(fcproj_w, fcproj_w_bf, (long)L * 4 * C * C);
    {   // lm_head: [C][V] f32 -> [V][C] bf16 (k-contiguous GEMM operand)
        dim3 g((V + 31) / 32, (C + 31) / 32);
        transpose_cvt_kernel<<<g, 256, 0, stream>>>(lm_w, lm_wT_bf, C, V);
    }

    auto gemm = [&](bool kcontig, int epi,
                    const bf16* A, long lda, const bf16* Bm, long ldbk, long ldbn,
                    const float* bias, const float* resid,
                    float* oF, bf16* oB, long ldc,
                    int Mm, int Nn, int Kk, float scale, int mask,
                    int batch, int zInner,
                    long aOut, long aIn, long bOut, long bIn, long cOut, long cIn) {
        dim3 grid((Nn + BN - 1) / BN, (Mm + BM - 1) / BM, batch);
        #define GEMM_CALL(KC, E)                                                 \
            wmma_gemm_kernel<KC, E><<<grid, 256, 0, stream>>>(                   \
                A, lda, Bm, ldbk, ldbn, bias, resid, oF, oB, ldc,                \
                Mm, Nn, Kk, scale, mask, zInner, aOut, aIn, bOut, bIn, cOut, cIn)
        if (kcontig) {
            GEMM_CALL(true, 0);                      // scores, lm_head
        } else {
            switch (epi) {
                case 1:  GEMM_CALL(false, 1); break; // proj, fcproj (+resid)
                case 2:  GEMM_CALL(false, 2); break; // qkv, att@V (bf16 out)
                default: GEMM_CALL(false, 3); break; // fc (gelu)
            }
        }
        #undef GEMM_CALL
    };

    embed_kernel<<<M, 256, 0, stream>>>(idx, wte, wpe, x, T, C);

    const float att_scale = 0.125f;  // 1/sqrt(D)
    for (int l = 0; l < L; ++l) {
        ln_bf16_kernel<<<M, 256, 0, stream>>>(x, ln1_w + (long)l * C,
                                              ln1_b + (long)l * C, ln_out, C);
        // qkv = ln_out @ attn_w + attn_b          (bf16 out)
        gemm(false, 2, ln_out, C, attn_w_bf + (long)l * C * 3 * C, 3 * C, 1,
             attn_b + (long)l * 3 * C, nullptr, nullptr, qkv, 3 * C,
             M, 3 * C, C, 1.0f, 0, 1, 1, 0, 0, 0, 0, 0, 0);
        // scores[b,h] = Q @ K^T * scale, causal   (f32 out; B k-contiguous)
        gemm(true, 0, qkv, 3 * C, qkv + C, 1, 3 * C,
             nullptr, nullptr, scores, nullptr, T,
             T, T, D, att_scale, 1,
             B * H, H,
             (long)T * 3 * C, (long)D,
             (long)T * 3 * C, (long)D,
             (long)H * T * T, (long)T * T);
        softmax_bf16_kernel<<<B * H * T, 256, 0, stream>>>(scores, probs, T);
        // y[b,h] = P @ V                           (bf16 out, strided heads)
        gemm(false, 2, probs, T, qkv + 2 * C, 3 * C, 1,
             nullptr, nullptr, nullptr, atty, C,
             T, D, T, 1.0f, 0,
             B * H, H,
             (long)H * T * T, (long)T * T,
             (long)T * 3 * C, (long)D,
             (long)T * C, (long)D);
        // x = x + atty @ proj_w + proj_b
        gemm(false, 1, atty, C, proj_w_bf + (long)l * C * C, C, 1,
             proj_b + (long)l * C, x, x, nullptr, C,
             M, C, C, 1.0f, 0, 1, 1, 0, 0, 0, 0, 0, 0);
        ln_bf16_kernel<<<M, 256, 0, stream>>>(x, ln2_w + (long)l * C,
                                              ln2_b + (long)l * C, ln_out, C);
        // fc_out = gelu(ln_out @ fc_w + fc_b)
        gemm(false, 3, ln_out, C, fc_w_bf + (long)l * C * 4 * C, 4 * C, 1,
             fc_b + (long)l * 4 * C, nullptr, nullptr, fc_out, 4 * C,
             M, 4 * C, C, 1.0f, 0, 1, 1, 0, 0, 0, 0, 0, 0);
        // x = x + fc_out @ fcproj_w + fcproj_b
        gemm(false, 1, fc_out, 4 * C, fcproj_w_bf + (long)l * 4 * C * C, C, 1,
             fcproj_b + (long)l * C, x, x, nullptr, C,
             M, C, 4 * C, 1.0f, 0, 1, 1, 0, 0, 0, 0, 0, 0);
    }

    ln_bf16_kernel<<<M, 256, 0, stream>>>(x, lnf_w, lnf_b, ln_out, C);
    // logits = ln_out @ lm_head  (B transposed in workspace -> k-contiguous)
    gemm(true, 0, ln_out, C, lm_wT_bf, 1, C,
         nullptr, nullptr, logits, nullptr, V,
         M, V, C, 1.0f, 0, 1, 1, 0, 0, 0, 0, 0, 0);
}
